// NeuralNetwork_53051436040900
// MI455X (gfx1250) — matrix operand order repaired and checked by reading, not compiled, not run on hardware
//
#include <hip/hip_runtime.h>
#include <hip/hip_bf16.h>
#include <stdint.h>
#include <stddef.h>

// Problem constants (from reference): VOCAB=50000, EMB=300, HID=1024, T=512, B=64
#define B_    64
#define T_    512
#define HID_  1024
#define EMB_  300
#define EMBP_ 320          // EMB padded to a multiple of 32 (WMMA K-step)
#define M_    (B_ * T_)    // 32768 rows for the parallel GEMMs
#define REC_BLOCKS 32      // persistent workgroups in the scan kernel

typedef __attribute__((ext_vector_type(16))) __bf16       v16bf;
typedef __attribute__((ext_vector_type(8)))  float        v8f;
typedef __attribute__((ext_vector_type(4)))  unsigned int v4u;

union FragU { v4u q[2]; v16bf v; };
static_assert(sizeof(v16bf) == 32, "v16bf must be 8 VGPRs");

__device__ __forceinline__ unsigned short f2bf(float f) {
  unsigned u = __builtin_bit_cast(unsigned, f);
  unsigned r = 0x7FFFu + ((u >> 16) & 1u);          // round-to-nearest-even
  return (unsigned short)((u + r) >> 16);
}

// A-fragment 16x32 bf16 (ISA 7.12.2): lanes 0-15 rows m0..m0+15 hold K{0..7,16..23},
// lanes 16-31 same rows hold K{8..15,24..31} -> two 16B loads per lane.
__device__ __forceinline__ v16bf load_frag_a(const unsigned short* base, size_t ld,
                                             int m0, int k, int lane) {
  const int row = m0 + (lane & 15);
  const int sel = (lane >> 4) * 8;
  const unsigned short* p = base + (size_t)row * ld + (size_t)k;
  FragU f;
  f.q[0] = *(const v4u*)(p + sel);
  f.q[1] = *(const v4u*)(p + 16 + sel);
  return f.v;
}

// B-fragment 32x16 bf16 taken from row-major W (N x K): B column n0+n == W row n0+n.
// Lanes 0-15 hold K{0..7,16..23}, lanes 16-31 hold K{8..15,24..31} -> two 16B loads per lane.
__device__ __forceinline__ v16bf load_frag_b(const unsigned short* base, size_t ld,
                                             int n0, int k, int lane) {
  const unsigned short* p = base + (size_t)(n0 + (lane & 15)) * ld
                                 + (size_t)k + (size_t)((lane >> 4) * 8);
  FragU f;
  f.q[0] = *(const v4u*)(p);
  f.q[1] = *(const v4u*)(p + 16);
  return f.v;
}

__device__ __forceinline__ v8f wmma_bf16(v16bf a, v16bf b, v8f c) {
  // (neg_a, A, neg_b, B, c_mod, C, reuse_a, reuse_b)
  return __builtin_amdgcn_wmma_f32_16x16x32_bf16(false, a, false, b, (short)0, c,
                                                 false, false);
}

// ---------------- prep kernels ----------------

__global__ void k_gather(const int* __restrict__ x, const float* __restrict__ emb,
                         unsigned short* __restrict__ E) {
  const int row = blockIdx.x;                 // row = b*T + t
  const int idx = x[row];
  const float* src = emb + (size_t)idx * EMB_;
  unsigned short* dst = E + (size_t)row * EMBP_;
  for (int c = threadIdx.x; c < EMBP_; c += blockDim.x)
    dst[c] = (c < EMB_) ? f2bf(src[c]) : (unsigned short)0;
}

__global__ void k_cvt_pad(const float* __restrict__ in, unsigned short* __restrict__ out,
                          int C, int Cpad) {
  const int r = blockIdx.x;
  for (int c = threadIdx.x; c < Cpad; c += blockDim.x)
    out[(size_t)r * Cpad + c] = (c < C) ? f2bf(in[(size_t)r * C + c]) : (unsigned short)0;
}

__global__ void k_addvec(const float* __restrict__ a, const float* __restrict__ b,
                         float* __restrict__ o, int n) {
  const int i = blockIdx.x * blockDim.x + threadIdx.x;
  if (i < n) o[i] = a[i] + b[i];
}

// ---------------- parallel GEMM:  C[M,N] = A(MxK bf16) * W(NxK bf16)^T + bias ----------
// block tile 256x64, 8 wave32s, one 32(M)x64(N) tile per wave (8 WMMAs / k-chunk).
// Ping-pong double-buffered fragments: even chunks in buffer A, odd in buffer B,
// so the steady state has zero register-rotation moves and loads overlap WMMAs.
// Requires K to be an even multiple of 32 (320 and 1024 both are).

__global__ void __launch_bounds__(256) k_gemm_bf16(
    const unsigned short* __restrict__ A,
    const unsigned short* __restrict__ W,
    const float* __restrict__ bias,
    float* __restrict__ C,
    int M, int N, int K) {
  const int nbn  = N / 64;
  const int bm   = blockIdx.x / nbn;
  const int bn   = blockIdx.x % nbn;
  const int wave = threadIdx.x >> 5;
  const int lane = threadIdx.x & 31;
  const int m0   = bm * 256 + wave * 32;
  const int n0   = bn * 64;
  (void)M;

  v8f acc[2][4];
#pragma unroll
  for (int i = 0; i < 2; ++i)
#pragma unroll
    for (int j = 0; j < 4; ++j)
#pragma unroll
      for (int r = 0; r < 8; ++r) acc[i][j][r] = 0.0f;

  v16bf a0A, a1A, bA[4];   // even-chunk buffers
  v16bf a0B, a1B, bB[4];   // odd-chunk buffers

  // prologue: chunk 0 -> buffer A
  a0A = load_frag_a(A, (size_t)K, m0, 0, lane);
  a1A = load_frag_a(A, (size_t)K, m0 + 16, 0, lane);
#pragma unroll
  for (int j = 0; j < 4; ++j) bA[j] = load_frag_b(W, (size_t)K, n0 + j * 16, 0, lane);

  const int KC = K >> 5;       // number of 32-wide chunks (even)
  int k = 32;
  for (int it = 0; it < KC - 2; it += 2) {
    // load odd chunk -> B while consuming A
    a0B = load_frag_a(A, (size_t)K, m0, k, lane);
    a1B = load_frag_a(A, (size_t)K, m0 + 16, k, lane);
#pragma unroll
    for (int j = 0; j < 4; ++j) bB[j] = load_frag_b(W, (size_t)K, n0 + j * 16, k, lane);
#pragma unroll
    for (int j = 0; j < 4; ++j) {
      acc[0][j] = wmma_bf16(a0A, bA[j], acc[0][j]);
      acc[1][j] = wmma_bf16(a1A, bA[j], acc[1][j]);
    }
    k += 32;
    // load even chunk -> A while consuming B
    a0A = load_frag_a(A, (size_t)K, m0, k, lane);
    a1A = load_frag_a(A, (size_t)K, m0 + 16, k, lane);
#pragma unroll
    for (int j = 0; j < 4; ++j) bA[j] = load_frag_b(W, (size_t)K, n0 + j * 16, k, lane);
#pragma unroll
    for (int j = 0; j < 4; ++j) {
      acc[0][j] = wmma_bf16(a0B, bB[j], acc[0][j]);
      acc[1][j] = wmma_bf16(a1B, bB[j], acc[1][j]);
    }
    k += 32;
  }
  // epilogue: last odd chunk -> B, then drain both buffers
  a0B = load_frag_a(A, (size_t)K, m0, k, lane);
  a1B = load_frag_a(A, (size_t)K, m0 + 16, k, lane);
#pragma unroll
  for (int j = 0; j < 4; ++j) bB[j] = load_frag_b(W, (size_t)K, n0 + j * 16, k, lane);
#pragma unroll
  for (int j = 0; j < 4; ++j) {
    acc[0][j] = wmma_bf16(a0A, bA[j], acc[0][j]);
    acc[1][j] = wmma_bf16(a1A, bA[j], acc[1][j]);
  }
#pragma unroll
  for (int j = 0; j < 4; ++j) {
    acc[0][j] = wmma_bf16(a0B, bB[j], acc[0][j]);
    acc[1][j] = wmma_bf16(a1B, bB[j], acc[1][j]);
  }

#pragma unroll
  for (int i = 0; i < 2; ++i) {
    const int mlo = m0 + i * 16 + (lane >> 4) * 8;  // C layout: VGPR r -> M=mlo+r, N=lane&15
#pragma unroll
    for (int j = 0; j < 4; ++j) {
      const int ncol = n0 + j * 16 + (lane & 15);
      const float bs = bias[ncol];
#pragma unroll
      for (int r = 0; r < 8; ++r)
        C[(size_t)(mlo + r) * N + ncol] = acc[i][j][r] + bs;
    }
  }
}

// ---------------- persistent recurrent scan ----------------
// 32 blocks; block owns 32 output features. W_hh slice (64KB bf16) is staged into
// LDS with gfx1250 async loads (GLOBAL_LOAD_ASYNC_TO_LDS_B128, ASYNCcnt-tracked).
// Each of 8 waves owns one 16(M)x16(N) tile of the 64x32 block output.
// Ping-pong K loop (HID/32 = 32 chunks, even). Device-wide barrier per timestep.

template <int STORE_SEQ>
__global__ void __launch_bounds__(256) k_rnn_scan(
    const float* __restrict__ pre,          // [B*T, HID], row = b*T + t, biases folded in
    const unsigned short* __restrict__ W,   // [HID, HID] bf16 (W_hh, row-major)
    unsigned short* __restrict__ h_ping,    // [B, HID] bf16
    unsigned short* __restrict__ h_pong,
    unsigned short* __restrict__ h_seq,     // [B*T, HID] bf16 (layer0) or unused
    float* __restrict__ h_final,            // [B, HID] f32 (layer1) or unused
    unsigned* __restrict__ bar_cnt,
    unsigned* __restrict__ bar_gen) {
  __shared__ unsigned short Wl[32 * HID_];  // 64 KB of the WGP's 320 KB LDS

  const int jbase = blockIdx.x * 32;
  {
    // rows jbase..jbase+31 of row-major W are one contiguous 64KB span:
    // copy with async global->LDS DMA, 16B per lane per op.
    const unsigned short* gsrc = W + (size_t)jbase * HID_;
    const unsigned lds_base = (unsigned)(uintptr_t)&Wl[0];   // LDS aperture: addr[31:0]
    for (int i = threadIdx.x; i < (32 * HID_) / 8; i += 256) {
      const unsigned boff = (unsigned)i * 16u;
      asm volatile("global_load_async_to_lds_b128 %0, %1, %2"
                   :: "v"(lds_base + boff), "v"(boff), "s"(gsrc)
                   : "memory");
    }
    asm volatile("s_wait_asynccnt 0x0" ::: "memory");
  }
  __syncthreads();

  const int wave = threadIdx.x >> 5;
  const int lane = threadIdx.x & 31;
  const int mt   = (wave & 3) * 16;         // batch-row tile base (0..48)
  const int nt   = (wave >> 2) * 16;        // feature tile base within slice (0/16)
  const int ncol = jbase + nt + (lane & 15);
  const int mlo  = mt + (lane >> 4) * 8;

  for (int t = 0; t < T_; ++t) {
    const unsigned short* hprev = (t & 1) ? h_pong : h_ping;
    unsigned short*       hnext = (t & 1) ? h_ping : h_pong;

    v8f acc;                                 // init with pre (includes b_ih + b_hh)
#pragma unroll
    for (int r = 0; r < 8; ++r)
      acc[r] = pre[((size_t)(mlo + r) * T_ + t) * HID_ + ncol];

    // prefetch next timestep's pre rows (cold 128MB stream) while we compute
    if (t + 1 < T_) {
#pragma unroll
      for (int r = 0; r < 8; ++r)
        __builtin_prefetch(&pre[((size_t)(mlo + r) * T_ + (t + 1)) * HID_ + ncol], 0, 1);
    }

    // ping-pong K loop: A from global (L2-hot 128KB), B from LDS. 32 chunks.
    v16bf aA = load_frag_a(hprev, HID_, mt, 0, lane);
    v16bf bA = load_frag_b(Wl, HID_, nt, 0, lane);
    v16bf aB, bB;
    int k = 32;
#pragma unroll 3
    for (int it = 0; it < (HID_ / 32 - 2) / 2; ++it) {   // 15 iterations
      aB = load_frag_a(hprev, HID_, mt, k, lane);
      bB = load_frag_b(Wl, HID_, nt, k, lane);
      acc = wmma_bf16(aA, bA, acc);
      k += 32;
      aA = load_frag_a(hprev, HID_, mt, k, lane);
      bA = load_frag_b(Wl, HID_, nt, k, lane);
      acc = wmma_bf16(aB, bB, acc);
      k += 32;
    }
    aB = load_frag_a(hprev, HID_, mt, k, lane);
    bB = load_frag_b(Wl, HID_, nt, k, lane);
    acc = wmma_bf16(aA, bA, acc);
    acc = wmma_bf16(aB, bB, acc);

#pragma unroll
    for (int r = 0; r < 8; ++r) {
      const float hv = tanhf(acc[r]);
      const int bidx = mlo + r;
      hnext[(size_t)bidx * HID_ + ncol] = f2bf(hv);
      if (STORE_SEQ)
        h_seq[((size_t)bidx * T_ + t) * HID_ + ncol] = f2bf(hv);
      if (!STORE_SEQ && t == T_ - 1)
        h_final[(size_t)bidx * HID_ + ncol] = hv;
    }

    // ---- device-wide barrier (release h_next, acquire for next step) ----
    __threadfence();
    __syncthreads();
    if (threadIdx.x == 0) {
      const unsigned g = __hip_atomic_load(bar_gen, __ATOMIC_ACQUIRE,
                                           __HIP_MEMORY_SCOPE_AGENT);
      const unsigned a = __hip_atomic_fetch_add(bar_cnt, 1u, __ATOMIC_ACQ_REL,
                                                __HIP_MEMORY_SCOPE_AGENT);
      if (a == (unsigned)gridDim.x - 1u) {
        __hip_atomic_store(bar_cnt, 0u, __ATOMIC_RELAXED, __HIP_MEMORY_SCOPE_AGENT);
        __hip_atomic_fetch_add(bar_gen, 1u, __ATOMIC_RELEASE, __HIP_MEMORY_SCOPE_AGENT);
      } else {
        while (__hip_atomic_load(bar_gen, __ATOMIC_ACQUIRE,
                                 __HIP_MEMORY_SCOPE_AGENT) == g)
          __builtin_amdgcn_s_sleep(1);
      }
    }
    __syncthreads();
    __threadfence();   // invalidate L0 so next step's hprev loads see fresh data
  }
}

// ---------------- final FC + sigmoid ----------------
__global__ void __launch_bounds__(256) k_fc_sigmoid(
    const float* __restrict__ hf, const float* __restrict__ Wfc,
    const float* __restrict__ bfc, float* __restrict__ out) {
  __shared__ float red[256];
  const int b = blockIdx.x;
  float s = 0.f;
  for (int k = threadIdx.x; k < HID_; k += 256)
    s += hf[(size_t)b * HID_ + k] * Wfc[k];
  red[threadIdx.x] = s;
  __syncthreads();
  for (int st = 128; st > 0; st >>= 1) {
    if (threadIdx.x < st) red[threadIdx.x] += red[threadIdx.x + st];
    __syncthreads();
  }
  if (threadIdx.x == 0) out[b] = 1.f / (1.f + __expf(-(red[0] + bfc[0])));
}

// ---------------- workspace layout (bytes) ----------------
static constexpr size_t OFF_E    = 0;                                  // 32768x320 bf16
static constexpr size_t OFF_WIH0 = OFF_E    + (size_t)M_ * EMBP_ * 2;  // 1024x320 bf16
static constexpr size_t OFF_WHH0 = OFF_WIH0 + (size_t)HID_ * EMBP_ * 2;
static constexpr size_t OFF_WIH1 = OFF_WHH0 + (size_t)HID_ * HID_ * 2;
static constexpr size_t OFF_WHH1 = OFF_WIH1 + (size_t)HID_ * HID_ * 2;
static constexpr size_t OFF_B0   = OFF_WHH1 + (size_t)HID_ * HID_ * 2;
static constexpr size_t OFF_B1   = OFF_B0   + (size_t)HID_ * 4;
static constexpr size_t OFF_PRE  = OFF_B1   + (size_t)HID_ * 4;        // 32768x1024 f32
static constexpr size_t OFF_H1   = OFF_PRE  + (size_t)M_ * HID_ * 4;   // 32768x1024 bf16
static constexpr size_t OFF_PING = OFF_H1   + (size_t)M_ * HID_ * 2;   // 64x1024 bf16
static constexpr size_t OFF_PONG = OFF_PING + (size_t)B_ * HID_ * 2;
static constexpr size_t OFF_HF   = OFF_PONG + (size_t)B_ * HID_ * 2;   // 64x1024 f32
static constexpr size_t OFF_BAR  = OFF_HF   + (size_t)B_ * HID_ * 4;

extern "C" void kernel_launch(void* const* d_in, const int* in_sizes, int n_in,
                              void* d_out, int out_size, void* d_ws, size_t ws_size,
                              hipStream_t stream) {
  (void)in_sizes; (void)n_in; (void)out_size; (void)ws_size;
  const int*   x     = (const int*)d_in[0];
  const float* emb   = (const float*)d_in[1];
  const float* W_ih0 = (const float*)d_in[2];
  const float* W_hh0 = (const float*)d_in[3];
  const float* b_ih0 = (const float*)d_in[4];
  const float* b_hh0 = (const float*)d_in[5];
  const float* W_ih1 = (const float*)d_in[6];
  const float* W_hh1 = (const float*)d_in[7];
  const float* b_ih1 = (const float*)d_in[8];
  const float* b_hh1 = (const float*)d_in[9];
  const float* W_fc  = (const float*)d_in[10];
  const float* b_fc  = (const float*)d_in[11];
  float* out = (float*)d_out;

  char* ws = (char*)d_ws;
  unsigned short* Ebf    = (unsigned short*)(ws + OFF_E);
  unsigned short* Wih0bf = (unsigned short*)(ws + OFF_WIH0);
  unsigned short* Whh0bf = (unsigned short*)(ws + OFF_WHH0);
  unsigned short* Wih1bf = (unsigned short*)(ws + OFF_WIH1);
  unsigned short* Whh1bf = (unsigned short*)(ws + OFF_WHH1);
  float*          bias0  = (float*)(ws + OFF_B0);
  float*          bias1  = (float*)(ws + OFF_B1);
  float*          pre    = (float*)(ws + OFF_PRE);
  unsigned short* h1bf   = (unsigned short*)(ws + OFF_H1);
  unsigned short* hping  = (unsigned short*)(ws + OFF_PING);
  unsigned short* hpong  = (unsigned short*)(ws + OFF_PONG);
  float*          hfinal = (float*)(ws + OFF_HF);
  unsigned*       barcnt = (unsigned*)(ws + OFF_BAR);
  unsigned*       bargen = (unsigned*)(ws + OFF_BAR + 4);

  // zero barrier + h0 state (graph-capture-safe async memsets)
  hipMemsetAsync(ws + OFF_BAR, 0, 256, stream);
  hipMemsetAsync(ws + OFF_PING, 0, (size_t)B_ * HID_ * 2 * 2, stream);

  // prep: gather+bf16-convert embeddings, bf16-convert weights, fold biases
  k_gather<<<M_, 64, 0, stream>>>(x, emb, Ebf);
  k_cvt_pad<<<HID_, 128, 0, stream>>>(W_ih0, Wih0bf, EMB_, EMBP_);
  k_cvt_pad<<<HID_, 128, 0, stream>>>(W_hh0, Whh0bf, HID_, HID_);
  k_cvt_pad<<<HID_, 128, 0, stream>>>(W_ih1, Wih1bf, HID_, HID_);
  k_cvt_pad<<<HID_, 128, 0, stream>>>(W_hh1, Whh1bf, HID_, HID_);
  k_addvec<<<(HID_ + 255) / 256, 256, 0, stream>>>(b_ih0, b_hh0, bias0, HID_);
  k_addvec<<<(HID_ + 255) / 256, 256, 0, stream>>>(b_ih1, b_hh1, bias1, HID_);

  const int gemm_grid = (M_ / 256) * (HID_ / 64);   // 2048 blocks

  // layer 0: pre0 = E * W_ih0^T + (b_ih0 + b_hh0), then persistent scan
  k_gemm_bf16<<<gemm_grid, 256, 0, stream>>>(Ebf, Wih0bf, bias0, pre, M_, HID_, EMBP_);
  k_rnn_scan<1><<<REC_BLOCKS, 256, 0, stream>>>(pre, Whh0bf, hping, hpong, h1bf,
                                                nullptr, barcnt, bargen);

  // layer 1: pre1 = h1 * W_ih1^T + (b_ih1 + b_hh1), then scan (final state only)
  k_gemm_bf16<<<gemm_grid, 256, 0, stream>>>(h1bf, Wih1bf, bias1, pre, M_, HID_, HID_);
  hipMemsetAsync(ws + OFF_BAR, 0, 256, stream);
  hipMemsetAsync(ws + OFF_PING, 0, (size_t)B_ * HID_ * 2 * 2, stream);
  k_rnn_scan<0><<<REC_BLOCKS, 256, 0, stream>>>(pre, Whh1bf, hping, hpong, nullptr,
                                                hfinal, barcnt, bargen);

  // head: sigmoid(h_last @ W_fc^T + b_fc)
  k_fc_sigmoid<<<B_, 256, 0, stream>>>(hfinal, W_fc, b_fc, out);
}
